// APPLayer_6339371729463
// MI455X (gfx1250) — matrix operand order repaired and checked by reading, not compiled, not run on hardware
//
#include <hip/hip_runtime.h>
#include <hip/hip_bf16.h>
#include <math.h>

// ---------------- problem constants (from reference) ----------------
constexpr int C_    = 128;
constexpr int HEADS_= 8;
constexpr int PJ_   = 17;
constexpr int NTOK_ = 486;            // B*T
constexpr int STOT_ = 1020;
constexpr int MROWS_= NTOK_ * PJ_;    // 8262 query rows
constexpr long MVAL_ = (long)NTOK_ * STOT_; // 495720 value rows

typedef __attribute__((ext_vector_type(16))) __bf16 bf16x16;
typedef __attribute__((ext_vector_type(8)))  float  f32x8;

// ---------------- weight repack: fp32 [K,N] -> bf16 WMMA-B fragments ----------------
// Layout: [K/32][N/16][lane 0..31][16 bf16]; lane holds column n = nt*16 + lane%16,
// k = kt*32 + (lane/16)*16 + i  (i = 0..15), the same k order as the A fragments built in gemm_bias.
__global__ __launch_bounds__(256) void repack_weight(const float* __restrict__ W,
                                                     __bf16* __restrict__ Wp,
                                                     int K, int N) {
    int tid = blockIdx.x * blockDim.x + threadIdx.x;
    int KT = K >> 5, NT = N >> 4;
    int total = KT * NT * 32;
    if (tid >= total) return;
    int lane = tid & 31;
    int nt   = (tid >> 5) % NT;
    int kt   = (tid >> 5) / NT;
    int n     = nt * 16 + (lane & 15);
    int kbase = kt * 32 + (lane >> 4) * 16;
    __bf16* dst = Wp + (size_t)tid * 16;
#pragma unroll
    for (int i = 0; i < 16; ++i)
        dst[i] = (__bf16)W[(size_t)(kbase + i) * N + n];
}

// ---------------- WMMA GEMM: Out[M,N] = act(A[M,K] @ W + bias) (+resid) ----------------
// 256 threads = 8 waves; each wave computes one 16xN strip; grid-stride over M tiles.
template<int K, int N, int ACT, bool RES>
__global__ __launch_bounds__(256) void gemm_bias(const float* __restrict__ A,
                                                 const __bf16* __restrict__ Wp,
                                                 const float* __restrict__ bias,
                                                 const float* __restrict__ resid,
                                                 float* __restrict__ Out, int M) {
    constexpr int KT = K / 32, NT = N / 16;
    const int wave = threadIdx.x >> 5;
    const int lane = threadIdx.x & 31;
    const int mtiles = (M + 15) >> 4;

    f32x8 acc[NT];
    for (int mt = blockIdx.x * 8 + wave; mt < mtiles; mt += gridDim.x * 8) {
#pragma unroll
        for (int i = 0; i < NT; ++i) acc[i] = (f32x8){0,0,0,0,0,0,0,0};

        // A fragment rows: lane<16 -> row lane, k {0..15}; lane>=16 -> row lane-16, k {16..31}
        int rowA = mt * 16 + (lane & 15);
        if (rowA >= M) rowA = M - 1;              // clamp; stores are guarded
        const float* arow = A + (size_t)rowA * K;
        const int klo = (lane >> 4) * 16;

#pragma unroll
        for (int kt = 0; kt < KT; ++kt) {
            bf16x16 afrag;
            const float* a0 = arow + kt * 32 + klo;
#pragma unroll
            for (int i = 0; i < 8; ++i) afrag[i]     = (__bf16)a0[i];
#pragma unroll
            for (int i = 0; i < 8; ++i) afrag[8 + i] = (__bf16)a0[8 + i];
#pragma unroll
            for (int nt = 0; nt < NT; ++nt) {
                bf16x16 bfrag = *(const bf16x16*)(Wp + ((size_t)(kt * NT + nt) * 32 + lane) * 16);
                acc[nt] = __builtin_amdgcn_wmma_f32_16x16x32_bf16(
                    false, afrag, false, bfrag, (short)0, acc[nt], false, false);
            }
        }

        // epilogue: D layout -> lane<16: rows 0..7 col lane; lane>=16: rows 8..15 col lane-16
        const int n0  = lane & 15;
        const int mlo = (lane >> 4) * 8;
#pragma unroll
        for (int nt = 0; nt < NT; ++nt) {
            const int n = nt * 16 + n0;
            const float bv = bias ? bias[n] : 0.0f;
#pragma unroll
            for (int r = 0; r < 8; ++r) {
                const int m = mt * 16 + mlo + r;
                if (m < M) {
                    float val = acc[nt][r] + bv;
                    if (ACT == 1) val = 0.5f * val * (1.0f + erff(val * 0.70710678118654752f));
                    if (RES) val += resid[(size_t)m * N + n];
                    Out[(size_t)m * N + n] = val;
                }
            }
        }
    }
}

// ---------------- LayerNorm over C=128: one wave per row ----------------
__global__ __launch_bounds__(256) void layernorm_k(const float* __restrict__ X,
                                                   const float* __restrict__ g,
                                                   const float* __restrict__ b,
                                                   float* __restrict__ Y, int M) {
    int wave = threadIdx.x >> 5, lane = threadIdx.x & 31;
    int row = blockIdx.x * (blockDim.x >> 5) + wave;
    if (row >= M) return;
    const float4 v = ((const float4*)(X + (size_t)row * C_))[lane];
    float s = v.x + v.y + v.z + v.w;
#pragma unroll
    for (int o = 16; o >= 1; o >>= 1) s += __shfl_xor(s, o, 32);
    const float mu = s * (1.0f / 128.0f);
    const float dx = v.x - mu, dy = v.y - mu, dz = v.z - mu, dw = v.w - mu;
    float s2 = dx * dx + dy * dy + dz * dz + dw * dw;
#pragma unroll
    for (int o = 16; o >= 1; o >>= 1) s2 += __shfl_xor(s2, o, 32);
    const float rstd = rsqrtf(s2 * (1.0f / 128.0f) + 1e-5f);
    const float4 gg = ((const float4*)g)[lane];
    const float4 bb = ((const float4*)b)[lane];
    float4 o4;
    o4.x = dx * rstd * gg.x + bb.x;
    o4.y = dy * rstd * gg.y + bb.y;
    o4.z = dz * rstd * gg.z + bb.z;
    o4.w = dw * rstd * gg.w + bb.w;
    ((float4*)(Y + (size_t)row * C_))[lane] = o4;
}

// ---------------- softmax over 16 (LEVELS*POINTS) per (row, head) ----------------
__global__ __launch_bounds__(256) void softmax16_k(const float* __restrict__ L,
                                                   float* __restrict__ P, int total) {
    int t = blockIdx.x * blockDim.x + threadIdx.x;
    if (t >= total) return;
    const float* in = L + (size_t)t * 16;
    float v[16], m = -INFINITY;
#pragma unroll
    for (int i = 0; i < 16; ++i) { v[i] = in[i]; m = fmaxf(m, v[i]); }
    float s = 0.0f;
#pragma unroll
    for (int i = 0; i < 16; ++i) { v[i] = __expf(v[i] - m); s += v[i]; }
    const float inv = 1.0f / s;
    float* out = P + (size_t)t * 16;
#pragma unroll
    for (int i = 0; i < 16; ++i) out[i] = v[i] * inv;
}

// ---------------- deformable bilinear sampling ----------------
// 16 lanes handle the d-slice of one (n, q, h) group; zero-padding out of bounds.
__global__ __launch_bounds__(256) void deform_sample_k(const float* __restrict__ V,   // [NTOK,STOT,128]
                                                       const float* __restrict__ off, // [MROWS,256] (h,l,p,2)
                                                       const float* __restrict__ aw,  // [MROWS,128] (h,l*p)
                                                       const float* __restrict__ ref, // [MROWS,8]   (l,2)
                                                       float* __restrict__ Out) {     // [MROWS,128]
    const int t = blockIdx.x * blockDim.x + threadIdx.x;
    const int dlane = t & 15;
    const int g = t >> 4;
    if (g >= MROWS_ * HEADS_) return;
    const int h  = g & 7;
    const int mq = g >> 3;
    const int n  = mq / PJ_;

    const int Hl[4] = {32, 16, 8, 4};
    const int Wl[4] = {24, 12, 6, 3};
    const int Sl[4] = {0, 768, 960, 1008};

    const float* offp = off + (size_t)mq * 256 + h * 32;
    const float* awp  = aw  + (size_t)mq * 128 + h * 16;
    const float* refp = ref + (size_t)mq * 8;
    const float* vb   = V + (size_t)n * STOT_ * C_ + h * 16 + dlane;

    float acc = 0.0f;
#pragma unroll
    for (int l = 0; l < 4; ++l) {
        const float rx = refp[l * 2 + 0], ry = refp[l * 2 + 1];
        const int Ww = Wl[l], Hh = Hl[l], s0 = Sl[l];
#pragma unroll
        for (int p = 0; p < 4; ++p) {
            const float ox = offp[(l * 4 + p) * 2 + 0];
            const float oy = offp[(l * 4 + p) * 2 + 1];
            // loc = ref + off/(W,H); grid coords: loc*W - 0.5
            const float lx = rx * (float)Ww + ox - 0.5f;
            const float ly = ry * (float)Hh + oy - 0.5f;
            const float x0f = floorf(lx), y0f = floorf(ly);
            const float fx = lx - x0f, fy = ly - y0f;
            const int x0 = (int)x0f, y0 = (int)y0f;
            float sample = 0.0f;
#pragma unroll
            for (int dy = 0; dy < 2; ++dy) {
#pragma unroll
                for (int dx = 0; dx < 2; ++dx) {
                    const int xi = x0 + dx, yi = y0 + dy;
                    if (xi >= 0 && xi < Ww && yi >= 0 && yi < Hh) {
                        const float wgt = (dx ? fx : 1.0f - fx) * (dy ? fy : 1.0f - fy);
                        sample += wgt * vb[(size_t)(s0 + yi * Ww + xi) * C_];
                    }
                }
            }
            acc += awp[l * 4 + p] * sample;
        }
    }
    Out[(size_t)mq * 128 + h * 16 + dlane] = acc;
}

// ---------------- small cross-attention: one thread per (n, h, q); Lk = PJ = 17 ----------------
__global__ __launch_bounds__(256) void cross_attn_k(const float* __restrict__ Q,
                                                    const float* __restrict__ Kk,
                                                    const float* __restrict__ Vv,
                                                    float* __restrict__ Out) {
    const int t = blockIdx.x * blockDim.x + threadIdx.x;
    if (t >= NTOK_ * HEADS_ * PJ_) return;
    const int qi = t % PJ_;
    const int h  = (t / PJ_) % HEADS_;
    const int n  = t / (PJ_ * HEADS_);

    float qv[16];
    const float* qp = Q + ((size_t)(n * PJ_ + qi) * C_) + h * 16;
#pragma unroll
    for (int d = 0; d < 16; ++d) qv[d] = qp[d];

    float sc[PJ_], mx = -INFINITY;
    for (int k = 0; k < PJ_; ++k) {
        const float* kp = Kk + ((size_t)(n * PJ_ + k) * C_) + h * 16;
        float s = 0.0f;
#pragma unroll
        for (int d = 0; d < 16; ++d) s += qv[d] * kp[d];
        s *= 0.25f;                       // 1/sqrt(16)
        sc[k] = s;
        mx = fmaxf(mx, s);
    }
    float sum = 0.0f;
    for (int k = 0; k < PJ_; ++k) { sc[k] = __expf(sc[k] - mx); sum += sc[k]; }
    const float inv = 1.0f / sum;

    float o[16];
#pragma unroll
    for (int d = 0; d < 16; ++d) o[d] = 0.0f;
    for (int k = 0; k < PJ_; ++k) {
        const float w = sc[k] * inv;
        const float* vp = Vv + ((size_t)(n * PJ_ + k) * C_) + h * 16;
#pragma unroll
        for (int d = 0; d < 16; ++d) o[d] += w * vp[d];
    }
    float* op = Out + ((size_t)(n * PJ_ + qi) * C_) + h * 16;
#pragma unroll
    for (int d = 0; d < 16; ++d) op[d] = o[d];
}

// ---------------- fat vectorized copy (f_maps passthrough output) ----------------
__global__ __launch_bounds__(256) void copy_f4(const float4* __restrict__ src,
                                               float4* __restrict__ dst, long n4) {
    long i = (long)blockIdx.x * blockDim.x + threadIdx.x;
    const long stride = (long)gridDim.x * blockDim.x;
    for (; i < n4; i += stride) dst[i] = src[i];
}

// ---------------- host orchestration ----------------
extern "C" void kernel_launch(void* const* d_in, const int* in_sizes, int n_in,
                              void* d_out, int out_size, void* d_ws, size_t ws_size,
                              hipStream_t stream) {
    (void)in_sizes; (void)n_in; (void)out_size; (void)ws_size;
    const float* x0    = (const float*)d_in[0];
    const float* x     = (const float*)d_in[1];
    const float* fmap  = (const float*)d_in[2];
    const float* refp  = (const float*)d_in[3];
    const float* ln1g  = (const float*)d_in[6];  const float* ln1b = (const float*)d_in[7];
    const float* w_off = (const float*)d_in[8];  const float* b_off= (const float*)d_in[9];
    const float* w_attn= (const float*)d_in[10]; const float* b_attn=(const float*)d_in[11];
    const float* w_val = (const float*)d_in[12]; const float* b_val= (const float*)d_in[13];
    const float* w_out = (const float*)d_in[14]; const float* b_out= (const float*)d_in[15];
    const float* ln2g  = (const float*)d_in[16]; const float* ln2b = (const float*)d_in[17];
    const float* wq    = (const float*)d_in[18]; const float* bq   = (const float*)d_in[19];
    const float* wk    = (const float*)d_in[20]; const float* bk   = (const float*)d_in[21];
    const float* wv    = (const float*)d_in[22]; const float* bv   = (const float*)d_in[23];
    const float* wco   = (const float*)d_in[24]; const float* bco  = (const float*)d_in[25];
    const float* ln3g  = (const float*)d_in[26]; const float* ln3b = (const float*)d_in[27];
    const float* wm1   = (const float*)d_in[28]; const float* bm1  = (const float*)d_in[29];
    const float* wm2   = (const float*)d_in[30]; const float* bm2  = (const float*)d_in[31];

    float* out_x = (float*)d_out;
    float* out_f = out_x + (size_t)MROWS_ * C_;   // f_maps region; used as temp V buffer first
    float* Vbuf  = out_f;

    size_t cur = 0;
    auto alloc = [&](size_t bytes) -> void* {
        void* p = (char*)d_ws + cur;
        cur += (bytes + 255) & ~(size_t)255;
        return p;
    };
    __bf16* wp_val = (__bf16*)alloc(128 * 128 * 2);
    __bf16* wp_off = (__bf16*)alloc(128 * 256 * 2);
    __bf16* wp_atn = (__bf16*)alloc(128 * 128 * 2);
    __bf16* wp_out = (__bf16*)alloc(128 * 128 * 2);
    __bf16* wp_q   = (__bf16*)alloc(128 * 128 * 2);
    __bf16* wp_k   = (__bf16*)alloc(128 * 128 * 2);
    __bf16* wp_v   = (__bf16*)alloc(128 * 128 * 2);
    __bf16* wp_co  = (__bf16*)alloc(128 * 128 * 2);
    __bf16* wp_m1  = (__bf16*)alloc(128 * 256 * 2);
    __bf16* wp_m2  = (__bf16*)alloc(256 * 128 * 2);
    float* xn     = (float*)alloc((size_t)MROWS_ * 128 * 4);
    float* offb   = (float*)alloc((size_t)MROWS_ * 256 * 4);
    float* logits = (float*)alloc((size_t)MROWS_ * 128 * 4);
    float* awb    = (float*)alloc((size_t)MROWS_ * 128 * 4);
    float* samp   = (float*)alloc((size_t)MROWS_ * 128 * 4);
    float* x1     = (float*)alloc((size_t)MROWS_ * 128 * 4);
    float* qb     = (float*)alloc((size_t)MROWS_ * 128 * 4);
    float* kb     = (float*)alloc((size_t)MROWS_ * 128 * 4);
    float* vb2    = (float*)alloc((size_t)MROWS_ * 128 * 4);
    float* attno  = (float*)alloc((size_t)MROWS_ * 128 * 4);
    float* x2     = (float*)alloc((size_t)MROWS_ * 128 * 4);
    float* hb     = (float*)alloc((size_t)MROWS_ * 256 * 4);

    auto cdiv = [](long a, long b) -> int { return (int)((a + b - 1) / b); };

    // 1) repack weights to bf16 WMMA-B fragment layout
    repack_weight<<<cdiv(1024, 256), 256, 0, stream>>>(w_val,  wp_val, 128, 128);
    repack_weight<<<cdiv(2048, 256), 256, 0, stream>>>(w_off,  wp_off, 128, 256);
    repack_weight<<<cdiv(1024, 256), 256, 0, stream>>>(w_attn, wp_atn, 128, 128);
    repack_weight<<<cdiv(1024, 256), 256, 0, stream>>>(w_out,  wp_out, 128, 128);
    repack_weight<<<cdiv(1024, 256), 256, 0, stream>>>(wq,     wp_q,   128, 128);
    repack_weight<<<cdiv(1024, 256), 256, 0, stream>>>(wk,     wp_k,   128, 128);
    repack_weight<<<cdiv(1024, 256), 256, 0, stream>>>(wv,     wp_v,   128, 128);
    repack_weight<<<cdiv(1024, 256), 256, 0, stream>>>(wco,    wp_co,  128, 128);
    repack_weight<<<cdiv(2048, 256), 256, 0, stream>>>(wm1,    wp_m1,  128, 256);
    repack_weight<<<cdiv(2048, 256), 256, 0, stream>>>(wm2,    wp_m2,  256, 128);

    const int mtv = (int)((MVAL_ + 15) / 16);   // 30983 value-row tiles
    const int mtr = (MROWS_ + 15) / 16;         // 517 query-row tiles
    const int gv  = cdiv(mtv, 8);
    const int gr  = cdiv(mtr, 8);
    const int gln = cdiv(MROWS_, 8);
    const int gnh = cdiv((long)MROWS_ * HEADS_, 256);

    // 2) value projection (dominant GEMM) -> V in out_f region (temp)
    gemm_bias<128, 128, 0, false><<<gv, 256, 0, stream>>>(fmap, wp_val, b_val, nullptr, Vbuf, (int)MVAL_);

    // 3) deformable self-attention branch
    layernorm_k<<<gln, 256, 0, stream>>>(x, ln1g, ln1b, xn, MROWS_);
    gemm_bias<128, 256, 0, false><<<gr, 256, 0, stream>>>(xn, wp_off, b_off, nullptr, offb, MROWS_);
    gemm_bias<128, 128, 0, false><<<gr, 256, 0, stream>>>(xn, wp_atn, b_attn, nullptr, logits, MROWS_);
    softmax16_k<<<gnh, 256, 0, stream>>>(logits, awb, MROWS_ * HEADS_);
    deform_sample_k<<<cdiv((long)MROWS_ * HEADS_ * 16, 256), 256, 0, stream>>>(Vbuf, offb, awb, refp, samp);
    gemm_bias<128, 128, 0, true><<<gr, 256, 0, stream>>>(samp, wp_out, b_out, x, x1, MROWS_);

    // 4) cross-attention to pose tokens
    layernorm_k<<<gln, 256, 0, stream>>>(x1, ln2g, ln2b, xn, MROWS_);
    gemm_bias<128, 128, 0, false><<<gr, 256, 0, stream>>>(xn, wp_q, bq, nullptr, qb, MROWS_);
    gemm_bias<128, 128, 0, false><<<gr, 256, 0, stream>>>(x0, wp_k, bk, nullptr, kb, MROWS_);
    gemm_bias<128, 128, 0, false><<<gr, 256, 0, stream>>>(x0, wp_v, bv, nullptr, vb2, MROWS_);
    cross_attn_k<<<gnh, 256, 0, stream>>>(qb, kb, vb2, attno);
    gemm_bias<128, 128, 0, true><<<gr, 256, 0, stream>>>(attno, wp_co, bco, x1, x2, MROWS_);

    // 5) MLP
    layernorm_k<<<gln, 256, 0, stream>>>(x2, ln3g, ln3b, xn, MROWS_);
    gemm_bias<128, 256, 1, false><<<gr, 256, 0, stream>>>(xn, wp_m1, bm1, nullptr, hb, MROWS_);
    gemm_bias<256, 128, 0, true><<<gr, 256, 0, stream>>>(hb, wp_m2, bm2, x2, out_x, MROWS_);

    // 6) f_maps passthrough output (overwrites the temp V buffer, after sampling consumed it)
    copy_f4<<<4096, 256, 0, stream>>>((const float4*)fmap, (float4*)out_f, MVAL_ * C_ / 4);
}
